// SAModule_20667382628496
// MI455X (gfx1250) — matrix operand order repaired and checked by reading, compiled, not run on hardware
//
#include <hip/hip_runtime.h>
#include <hip/hip_bf16.h>
#include <stdint.h>

#define B_GR 8
#define N_PTS 4096
#define M_CTR 2048
#define K_NBR 64
#define FIN 64
#define R2 0.25f
#define CPB 8   // centers per block in the WMMA MLP kernel

typedef _Float16 v16h __attribute__((ext_vector_type(16)));
typedef _Float16 h8   __attribute__((ext_vector_type(8)));
typedef _Float16 h4   __attribute__((ext_vector_type(4)));
typedef float    v8f  __attribute__((ext_vector_type(8)));
typedef int      vi4  __attribute__((vector_size(16)));   // raw clang vector, matches builtin param

#if defined(__has_builtin)
#if __has_builtin(__builtin_amdgcn_global_load_async_to_lds_b128)
#define HAVE_ASYNC_LDS 1
#endif
#endif

// ---------- monotone float <-> uint key for atomic max ----------
__device__ inline unsigned int fkey(float f) {
  unsigned int b = __float_as_uint(f);
  return (b & 0x80000000u) ? ~b : (b | 0x80000000u);
}
__device__ inline float fdec(unsigned int k) {
  unsigned int b = (k & 0x80000000u) ? (k & 0x7fffffffu) : ~k;
  return __uint_as_float(b);
}

// ---------- WMMA fragment loaders (16x16x32 f16, wave32 layouts) ----------
// A 16x32 tile from row-major f16 [rows x stride]; per-lane data is two
// contiguous 16B runs (K=kb..kb+7 and K=kb+16..kb+23) -> 2x ds_load_b128.
__device__ inline v16h loadA(const _Float16* src, int stride, int mbase, int kc, int lane) {
  const _Float16* base = src + (mbase + (lane & 15)) * stride + kc + ((lane & 16) ? 8 : 0);
  h8 lo = *(const h8*)(base);
  h8 hi = *(const h8*)(base + 16);
  return __builtin_shufflevector(lo, hi, 0, 1, 2, 3, 4, 5, 6, 7, 8, 9, 10, 11, 12, 13, 14, 15);
}
// B 32x16 tile from TRANSPOSED weights wT[N x Kd]; per-lane data is two
// contiguous 16B runs (K=kb..kb+7 and K=kb+16..kb+23) -> 2x ds_load_b128.
__device__ inline v16h loadBT(const _Float16* srcT, int Kd, int nbase, int kc, int lane) {
  const _Float16* base = srcT + (nbase + (lane & 15)) * Kd + kc + ((lane & 16) ? 8 : 0);
  h8 lo = *(const h8*)(base);
  h8 hi = *(const h8*)(base + 16);
  return __builtin_shufflevector(lo, hi, 0, 1, 2, 3, 4, 5, 6, 7, 8, 9, 10, 11, 12, 13, 14, 15);
}

// ---------- kernel 0: pack weights to f16, TRANSPOSED, w1 K padded 67->96 ----------
// layout: wT1[64][96] | wT2[64][64] | wT3[128][64]   (N-major, K contiguous)
__global__ void pack_weights(const float* __restrict__ w1, const float* __restrict__ w2,
                             const float* __restrict__ w3, _Float16* __restrict__ wpack) {
  int idx = blockIdx.x * blockDim.x + threadIdx.x;
  if (idx < 6144) {                       // wT1: n in [0,64), k in [0,96)
    int n = idx / 96, k = idx - n * 96;
    wpack[idx] = (_Float16)(k < 67 ? w1[k * 64 + n] : 0.0f);
  } else if (idx < 10240) {               // wT2: n in [0,64), k in [0,64)
    int t = idx - 6144;
    int n = t >> 6, k = t & 63;
    wpack[idx] = (_Float16)w2[k * 64 + n];
  } else if (idx < 18432) {               // wT3: n in [0,128), k in [0,64)
    int t = idx - 10240;
    int n = t >> 6, k = t & 63;
    wpack[idx] = (_Float16)w3[k * 128 + n];
  }
}

// ---------- kernel 1: farthest point sampling (one WG per graph) ----------
__global__ __launch_bounds__(1024)
void fps_kernel(const float* __restrict__ pos, int* __restrict__ cidx,
                float* __restrict__ posc) {
  const int b   = blockIdx.x;
  const int tid = threadIdx.x;
  const float* pg = pos + (size_t)b * N_PTS * 3;

  float px[4], py[4], pz[4], mind[4];
#pragma unroll
  for (int t = 0; t < 4; ++t) {
    int i = tid * 4 + t;
    px[t] = pg[i * 3 + 0]; py[t] = pg[i * 3 + 1]; pz[t] = pg[i * 3 + 2];
    mind[t] = 1e30f;
  }
  __shared__ float s_wv[32];
  __shared__ int   s_wi[32];
  __shared__ int   s_last;
  __shared__ float s_lx, s_ly, s_lz;
  if (tid == 0) { s_last = 0; s_lx = pg[0]; s_ly = pg[1]; s_lz = pg[2]; }
  __syncthreads();

  for (int m = 0; m < M_CTR; ++m) {
    int   last = s_last;
    float lx = s_lx, ly = s_ly, lz = s_lz;
    if (tid == 0) {
      cidx[b * M_CTR + m] = last;
      posc[((size_t)b * M_CTR + m) * 3 + 0] = lx;
      posc[((size_t)b * M_CTR + m) * 3 + 1] = ly;
      posc[((size_t)b * M_CTR + m) * 3 + 2] = lz;
    }
    float bv = -1.0f; int bi = 0;
#pragma unroll
    for (int t = 0; t < 4; ++t) {
      float dx = px[t] - lx, dy = py[t] - ly, dz = pz[t] - lz;
      float d = dx * dx + dy * dy + dz * dz;
      mind[t] = fminf(mind[t], d);
      if (mind[t] > bv) { bv = mind[t]; bi = tid * 4 + t; }
    }
    // wave32 argmax reduce, ties -> lowest index (matches jnp.argmax)
#pragma unroll
    for (int off = 16; off > 0; off >>= 1) {
      float ov = __shfl_xor(bv, off, 32);
      int   oi = __shfl_xor(bi, off, 32);
      if (ov > bv || (ov == bv && oi < bi)) { bv = ov; bi = oi; }
    }
    if ((tid & 31) == 0) { s_wv[tid >> 5] = bv; s_wi[tid >> 5] = bi; }
    __syncthreads();
    if (tid == 0) {
      float v = s_wv[0]; int i = s_wi[0];
      for (int w = 1; w < 32; ++w)
        if (s_wv[w] > v || (s_wv[w] == v && s_wi[w] < i)) { v = s_wv[w]; i = s_wi[w]; }
      s_last = i;
      s_lx = pg[i * 3 + 0]; s_ly = pg[i * 3 + 1]; s_lz = pg[i * 3 + 2];
    }
    __syncthreads();
  }
}

// ---------- kernel 2: exact top-K=64 nearest (one WG per center) ----------
__global__ __launch_bounds__(128)
void knn_kernel(const float* __restrict__ pos, const float* __restrict__ posc,
                int* __restrict__ nbr) {
  const int c   = blockIdx.x;
  const int b   = c / M_CTR;
  const int tid = threadIdx.x;
  __shared__ float d2s[N_PTS];
  __shared__ float s_wv[4];
  __shared__ int   s_wi[4];
  __shared__ int   s_win;

  const float cx = posc[(size_t)c * 3 + 0];
  const float cy = posc[(size_t)c * 3 + 1];
  const float cz = posc[(size_t)c * 3 + 2];
  const float* pg = pos + (size_t)b * N_PTS * 3;

  for (int i = tid; i < N_PTS; i += 128) {
    float dx = pg[i * 3 + 0] - cx, dy = pg[i * 3 + 1] - cy, dz = pg[i * 3 + 2] - cz;
    d2s[i] = dx * dx + dy * dy + dz * dz;
  }
  __syncthreads();

  // cached per-thread segment min (32 elements each)
  float sm = 1e30f; int si = tid * 32;
  for (int t = 0; t < 32; ++t) {
    int i = tid * 32 + t; float v = d2s[i];
    if (v < sm) { sm = v; si = i; }
  }
  for (int k = 0; k < K_NBR; ++k) {
    float bv = sm; int bi = si;
#pragma unroll
    for (int off = 16; off > 0; off >>= 1) {
      float ov = __shfl_xor(bv, off, 32);
      int   oi = __shfl_xor(bi, off, 32);
      if (ov < bv || (ov == bv && oi < bi)) { bv = ov; bi = oi; }
    }
    if ((tid & 31) == 0) { s_wv[tid >> 5] = bv; s_wi[tid >> 5] = bi; }
    __syncthreads();
    if (tid == 0) {
      float v = s_wv[0]; int i = s_wi[0];
      for (int w = 1; w < 4; ++w)
        if (s_wv[w] < v || (s_wv[w] == v && s_wi[w] < i)) { v = s_wv[w]; i = s_wi[w]; }
      s_win = i;
      nbr[(size_t)c * K_NBR + k] = b * N_PTS + i;
    }
    __syncthreads();
    int wi = s_win;
    if ((wi >> 5) == tid) {          // only the owning segment re-scans
      d2s[wi] = 1e30f;
      sm = 1e30f; si = tid * 32;
      for (int t = 0; t < 32; ++t) {
        int i = tid * 32 + t; float v = d2s[i];
        if (v < sm) { sm = v; si = i; }
      }
    }
    __syncthreads();
  }
}

// ---------- kernel 3: PointConv MLP via WMMA + masked max-aggregate ----------
__global__ __launch_bounds__(128)
void pointconv_wmma(const float* __restrict__ x, const float* __restrict__ pos,
                    const float* __restrict__ posc, const int* __restrict__ nbr,
                    const _Float16* __restrict__ wpack,
                    const float* __restrict__ b1, const float* __restrict__ b2,
                    const float* __restrict__ b3, float* __restrict__ out_x) {
  __shared__ _Float16 wLDS[18432];     // wT1[64x96] | wT2[64x64] | wT3[128x64]
  __shared__ _Float16 h0[64 * 96];     // layer-0 input; reused as layer-2 output (stride 64)
  __shared__ _Float16 h1[64 * 64];
  __shared__ unsigned int soutKey[128];
  __shared__ int s_nbr[64];
  __shared__ int valids[64];

  const int tid  = threadIdx.x;
  const int lane = tid & 31;
  const int wv   = tid >> 5;           // wave id == M-tile (rows of neighbors)

  __builtin_prefetch(wpack, 0, 3);
  { // weight load into LDS: prefer async (ASYNCcnt) global->LDS path
    const vi4* src = (const vi4*)wpack;
    vi4* dst = (vi4*)wLDS;
#ifdef HAVE_ASYNC_LDS
    for (int i = tid; i < 2304; i += 128)
      __builtin_amdgcn_global_load_async_to_lds_b128((vi4*)(src + i), dst + i, 0, 0);
#if __has_builtin(__builtin_amdgcn_s_wait_asynccnt)
    __builtin_amdgcn_s_wait_asynccnt(0);
#else
    asm volatile("s_wait_asynccnt 0x0" ::: "memory");
#endif
#else
    for (int i = tid; i < 2304; i += 128) dst[i] = src[i];
#endif
  }
  const _Float16* w1T = wLDS;           // [64][96]
  const _Float16* w2T = wLDS + 6144;    // [64][64]
  const _Float16* w3T = wLDS + 10240;   // [128][64]
  const int mbase = wv * 16;
  const int rbase = mbase + ((lane & 16) ? 8 : 0);

  // hoist per-lane biases (constant across the center loop)
  float b1c[4], b2c[4];
#pragma unroll
  for (int nt = 0; nt < 4; ++nt) {
    b1c[nt] = b1[nt * 16 + (lane & 15)];
    b2c[nt] = b2[nt * 16 + (lane & 15)];
  }
  const float bias3 = b3[tid];

  for (int cc = 0; cc < CPB; ++cc) {
    const int c = blockIdx.x * CPB + cc;
    const float cx = posc[(size_t)c * 3 + 0];
    const float cy = posc[(size_t)c * 3 + 1];
    const float cz = posc[(size_t)c * 3 + 2];
    __syncthreads();                    // previous center fully consumed
    soutKey[tid] = 0u;                  // key of "most negative"
    if (tid < 64) s_nbr[tid] = nbr[(size_t)c * K_NBR + tid];
    __syncthreads();

    // gather h0 = [x_j | pos_j - pos_c | 0-pad] (64 x 96), float4 -> packed f16
    for (int s = tid; s < 64 * 16; s += 128) {
      int row = s >> 4, c4 = s & 15;
      int j = s_nbr[row];
      float4 v = ((const float4*)x)[(size_t)j * 16 + c4];
      h4 hv; hv[0] = (_Float16)v.x; hv[1] = (_Float16)v.y;
      hv[2] = (_Float16)v.z; hv[3] = (_Float16)v.w;
      *(h4*)(h0 + row * 96 + c4 * 4) = hv;
    }
    for (int s = tid; s < 64 * 8; s += 128) {   // cols 64..95: rel | zero pad
      int row = s >> 3, g = s & 7;
      h4 hv = {};
      if (g == 0) {
        int j = s_nbr[row];
        float dx = pos[(size_t)j * 3 + 0] - cx;
        float dy = pos[(size_t)j * 3 + 1] - cy;
        float dz = pos[(size_t)j * 3 + 2] - cz;
        hv[0] = (_Float16)dx; hv[1] = (_Float16)dy; hv[2] = (_Float16)dz;
        valids[row] = (dx * dx + dy * dy + dz * dz) <= R2 ? 1 : 0;
      }
      *(h4*)(h0 + row * 96 + 64 + g * 4) = hv;
    }
    __syncthreads();

    // ---- layer 1: relu(h0[64x96] @ w1[96x64] + b1) -> h1
    {
      v8f acc[4] = {};
#pragma unroll
      for (int kc = 0; kc < 96; kc += 32) {
        v16h a = loadA(h0, 96, mbase, kc, lane);
#pragma unroll
        for (int nt = 0; nt < 4; ++nt) {
          v16h bf = loadBT(w1T, 96, nt * 16, kc, lane);
          acc[nt] = __builtin_amdgcn_wmma_f32_16x16x32_f16(false, a, false, bf,
                                                           (short)0, acc[nt], false, false);
        }
      }
#pragma unroll
      for (int nt = 0; nt < 4; ++nt) {
        int col = nt * 16 + (lane & 15);
#pragma unroll
        for (int v = 0; v < 8; ++v)
          h1[(rbase + v) * 64 + col] = (_Float16)fmaxf(acc[nt][v] + b1c[nt], 0.0f);
      }
    }
    __syncthreads();

    // ---- layer 2: relu(h1 @ w2 + b2) -> h0 (reused, stride 64)
    {
      v8f acc[4] = {};
#pragma unroll
      for (int kc = 0; kc < 64; kc += 32) {
        v16h a = loadA(h1, 64, mbase, kc, lane);
#pragma unroll
        for (int nt = 0; nt < 4; ++nt) {
          v16h bf = loadBT(w2T, 64, nt * 16, kc, lane);
          acc[nt] = __builtin_amdgcn_wmma_f32_16x16x32_f16(false, a, false, bf,
                                                           (short)0, acc[nt], false, false);
        }
      }
#pragma unroll
      for (int nt = 0; nt < 4; ++nt) {
        int col = nt * 16 + (lane & 15);
#pragma unroll
        for (int v = 0; v < 8; ++v)
          h0[(rbase + v) * 64 + col] = (_Float16)fmaxf(acc[nt][v] + b2c[nt], 0.0f);
      }
    }
    __syncthreads();

    // ---- layer 3: h2 @ w3 [64x128]; masked max straight from accumulators
    {
      v16h a0 = loadA(h0, 64, mbase, 0, lane);
      v16h a1 = loadA(h0, 64, mbase, 32, lane);
#pragma unroll
      for (int nth = 0; nth < 2; ++nth) {
        v8f acc[4] = {};
#pragma unroll
        for (int nt = 0; nt < 4; ++nt) {
          int nb = (nth * 4 + nt) * 16;
          v16h bf0 = loadBT(w3T, 64, nb, 0, lane);
          acc[nt] = __builtin_amdgcn_wmma_f32_16x16x32_f16(false, a0, false, bf0,
                                                           (short)0, acc[nt], false, false);
          v16h bf1 = loadBT(w3T, 64, nb, 32, lane);
          acc[nt] = __builtin_amdgcn_wmma_f32_16x16x32_f16(false, a1, false, bf1,
                                                           (short)0, acc[nt], false, false);
        }
#pragma unroll
        for (int nt = 0; nt < 4; ++nt) {
          int col = (nth * 4 + nt) * 16 + (lane & 15);
#pragma unroll
          for (int v = 0; v < 8; ++v)
            if (valids[rbase + v]) atomicMax(&soutKey[col], fkey(acc[nt][v]));
        }
      }
    }
    __syncthreads();
    out_x[(size_t)c * 128 + tid] = fdec(soutKey[tid]) + bias3;  // bias after max
  }
}

// ---------- kernel 4: out_pos / out_batch tail ----------
__global__ void tail_kernel(const float* __restrict__ posc, float* __restrict__ out_pos,
                            float* __restrict__ out_batch) {
  int i = blockIdx.x * blockDim.x + threadIdx.x;
  if (i < B_GR * M_CTR * 3) out_pos[i] = posc[i];
  if (i < B_GR * M_CTR)     out_batch[i] = (float)(i / M_CTR);
}

extern "C" void kernel_launch(void* const* d_in, const int* in_sizes, int n_in,
                              void* d_out, int out_size, void* d_ws, size_t ws_size,
                              hipStream_t stream) {
  (void)in_sizes; (void)n_in; (void)out_size; (void)ws_size;
  const float* x   = (const float*)d_in[0];
  const float* pos = (const float*)d_in[1];
  // d_in[2] = batch (implied by equal-size graphs; unused)
  const float* w1 = (const float*)d_in[3];
  const float* b1 = (const float*)d_in[4];
  const float* w2 = (const float*)d_in[5];
  const float* b2 = (const float*)d_in[6];
  const float* w3 = (const float*)d_in[7];
  const float* b3 = (const float*)d_in[8];
  float* out = (float*)d_out;

  char* ws = (char*)d_ws;
  _Float16* wpack = (_Float16*)ws;                               // 36864 B
  int*      cidx  = (int*)(ws + 36864);                          // 65536 B
  float*    posc  = (float*)(ws + 36864 + 65536);                // 196608 B
  int*      nbr   = (int*)(ws + 36864 + 65536 + 196608);         // 4 MiB

  pack_weights<<<(18432 + 255) / 256, 256, 0, stream>>>(w1, w2, w3, wpack);
  fps_kernel<<<B_GR, 1024, 0, stream>>>(pos, cidx, posc);
  knn_kernel<<<B_GR * M_CTR, 128, 0, stream>>>(pos, posc, nbr);
  pointconv_wmma<<<(B_GR * M_CTR) / CPB, 128, 0, stream>>>(x, pos, posc, nbr, wpack,
                                                           b1, b2, b3, out);
  float* out_pos   = out + (size_t)B_GR * M_CTR * 128;
  float* out_batch = out_pos + (size_t)B_GR * M_CTR * 3;
  tail_kernel<<<(B_GR * M_CTR * 3 + 255) / 256, 256, 0, stream>>>(posc, out_pos, out_batch);
}
